// DGCNN_semseg_63668595196424
// MI455X (gfx1250) — matrix operand order repaired and checked by reading, not compiled, not run on hardware
//
#include <hip/hip_runtime.h>
#include <hip/hip_bf16.h>

#define BB 4
#define NN 4096
#define KK 32
#define BN (BB * NN)          // 16384 points total
#define BNK ((long)BN * KK)   // 524288 edge rows

typedef __attribute__((ext_vector_type(16))) _Float16 v16h;
typedef __attribute__((ext_vector_type(8)))  _Float16 v8h;
typedef __attribute__((ext_vector_type(8)))  float    v8f;
typedef _Float16 h16;

// ---------------------------------------------------------------------------
// WMMA fragment loaders (contiguous k = 16h + s order, same for A and B, wave32)
// A 16x32 (MxK), row-major source, stride lda (halfs, multiple of 8):
//   lanes 0-15 : M=lane,    K = 0..15 (16 contiguous halfs)
//   lanes16-31 : M=lane-16, K = 16..31
// ---------------------------------------------------------------------------
__device__ __forceinline__ v16h load_a_frag(const h16* A, int lda, int lane) {
  int m  = lane & 15;
  int hl = lane >> 4;
  const h16* p = A + (size_t)m * lda + hl * 16;
  v8h a0 = *(const v8h*)(p);
  v8h a1 = *(const v8h*)(p + 8);
  v16h r;
#pragma unroll
  for (int i = 0; i < 8; ++i) { r[i] = a0[i]; r[8 + i] = a1[i]; }
  return r;
}

// B 32x16 (KxN) where source is W(N_out, K) row-major (so B = W^T):
//   lanes 0-15 : col n=lane,    K = 0..15 (16 contiguous halfs of W row)
//   lanes16-31 : col n=lane-16, K = 16..31
__device__ __forceinline__ v16h load_b_frag(const h16* W, int ldw, int lane) {
  int n  = lane & 15;
  int hl = lane >> 4;
  const h16* p = W + (size_t)n * ldw + hl * 16;
  v8h b0 = *(const v8h*)(p);
  v8h b1 = *(const v8h*)(p + 8);
  v16h r;
#pragma unroll
  for (int i = 0; i < 8; ++i) { r[i] = b0[i]; r[8 + i] = b1[i]; }
  return r;
}

// Async global->LDS 16-byte copy (CDNA5 GLOBAL_LOAD_ASYNC_TO_LDS_B128,
// GV addressing: per-lane LDS byte offset in VDST vgpr, 64-bit addr in vgpr
// pair). Tracked by ASYNCcnt.
__device__ __forceinline__ void async_copy_b128(unsigned lds_off,
                                                const void* gaddr) {
  asm volatile("global_load_async_to_lds_b128 %0, %1, off"
               :: "v"(lds_off), "v"((unsigned long long)(uintptr_t)gaddr)
               : "memory");
}
__device__ __forceinline__ void wait_async0() {
  asm volatile("s_wait_asynccnt 0x0" ::: "memory");
}

// ---------------------------------------------------------------------------
// Generic GEMM: out(M,O) = leaky(A(M,K) * W(O,K)^T * s + b), f16 in/out,
// f32 accumulate via v_wmma_f32_16x16x32_f16.
// One block = one 16-wide N-tile x 8 M-tiles (one per wave). The 16xK weight
// panel (<= 38 KB for w7) is staged into LDS once per block with async
// global->LDS loads, then all 8 waves read B-fragments from LDS.
// ---------------------------------------------------------------------------
__global__ __launch_bounds__(256) void gemm_kernel(
    const h16* __restrict__ A, long M, int lda, int kSteps,
    const h16* __restrict__ W, int ldw,
    const float* __restrict__ scale, const float* __restrict__ bias,
    h16* __restrict__ out, int ldo, int O, int act) {
  __shared__ h16 sW[16 * 1280];  // weight panel, up to K=1216 -> 40 KB
  int  ntiles = O >> 4;
  int  nt   = blockIdx.x % ntiles;
  long mblk = blockIdx.x / ntiles;
  int  wave = threadIdx.x >> 5;
  int  lane = threadIdx.x & 31;
  int  Kd   = kSteps * 32;

  // ---- stage weight panel: 16 rows x Kd halfs, row stride Kd in LDS
  {
    const h16* Wt = W + (size_t)nt * 16 * ldw;
    unsigned  sbase  = (unsigned)(uintptr_t)(&sW[0]);
    int       chunks = 2 * Kd;                    // 16B chunks = 16*Kd*2/16
    for (int c = threadIdx.x; c < chunks; c += 256) {
      int ho  = c * 8;                            // half offset in panel
      int row = ho / Kd;
      int col = ho - row * Kd;
      async_copy_b128(sbase + (unsigned)c * 16u,
                      Wt + (size_t)row * ldw + col);
    }
  }
  wait_async0();
  __syncthreads();

  long mt = mblk * 8 + wave;
  if (mt * 16 >= M) return;
  const h16* Ab = A + (size_t)mt * 16 * lda;

  v8f acc = {};
  int n16 = lane & 15;
  int hl  = lane >> 4;
  for (int ks = 0; ks < kSteps; ++ks) {
    v16h af = load_a_frag(Ab + ks * 32, lda, lane);
    const h16* pb = &sW[n16 * Kd + ks * 32 + hl * 16];
    v8h b0 = *(const v8h*)(pb);
    v8h b1 = *(const v8h*)(pb + 8);
    v16h bf;
#pragma unroll
    for (int i = 0; i < 8; ++i) { bf[i] = b0[i]; bf[8 + i] = b1[i]; }
    acc = __builtin_amdgcn_wmma_f32_16x16x32_f16(false, af, false, bf,
                                                 (short)0, acc, false, false);
  }
  // C/D layout: VGPR v, lanes0-15 -> M=v, lanes16-31 -> M=8+v; N=lane&15
  int n = nt * 16 + n16;
  float s  = scale[n];
  float bo = bias[n];
#pragma unroll
  for (int v = 0; v < 8; ++v) {
    long m = mt * 16 + v + 8 * hl;
    float xv = acc[v] * s + bo;
    if (act) xv = (xv >= 0.f) ? xv : 0.2f * xv;
    out[(size_t)m * ldo + n] = (h16)xv;
  }
}

// ---------------------------------------------------------------------------
// Fused KNN: per 16-row tile, compute full 16x4096 stripe of
// pd = 2*<x_r,x_c> - xx_r - xx_c via WMMA into LDS (f16), then top-32 per row.
// A-fragments are loop-invariant and held in registers. Each LDS distance
// slot col is owned by lane (col & 31) -> no cross-lane hazards in selection.
// ---------------------------------------------------------------------------
__global__ __launch_bounds__(256) void knn_kernel(
    const h16* __restrict__ F, int ldf, int kSteps,
    const float* __restrict__ xx, int* __restrict__ knn_idx) {
  __shared__ h16 dist[16 * NN];  // 128 KB of the WGP's 320 KB LDS
  int b    = blockIdx.x / (NN / 16);
  int rt   = blockIdx.x % (NN / 16);
  int row0 = rt * 16;
  int wave = threadIdx.x >> 5;
  int lane = threadIdx.x & 31;
  const h16*   Fb  = F + (size_t)b * NN * ldf;
  const float* xxb = xx + b * NN;

  v16h afr[2];
  for (int ks = 0; ks < kSteps; ++ks)
    afr[ks] = load_a_frag(Fb + (size_t)row0 * ldf + ks * 32, ldf, lane);

  for (int ct = wave; ct < NN / 16; ct += 8) {
    v8f acc = {};
    for (int ks = 0; ks < kSteps; ++ks) {
      v16h bf = load_b_frag(Fb + (size_t)ct * 16 * ldf + ks * 32, ldf, lane);
      acc = __builtin_amdgcn_wmma_f32_16x16x32_f16(false, afr[ks], false, bf,
                                                   (short)0, acc, false, false);
    }
    int col = ct * 16 + (lane & 15);
    int hl  = lane >> 4;
    float xc = xxb[col];
#pragma unroll
    for (int v = 0; v < 8; ++v) {
      int m = v + 8 * hl;
      float pd = 2.f * acc[v] - xxb[row0 + m] - xc;
      dist[m * NN + col] = (h16)pd;
    }
  }
  __syncthreads();

  for (int rr = wave; rr < 16; rr += 8) {
    h16* drow = dist + rr * NN;
    int* orow = knn_idx + ((size_t)(b * NN + row0 + rr)) * KK;
    for (int k = 0; k < KK; ++k) {
      float best = -3.0e38f;
      int   bi   = lane;
      for (int c = lane; c < NN; c += 32) {
        float v = (float)drow[c];
        if (v > best) { best = v; bi = c; }
      }
#pragma unroll
      for (int off = 16; off > 0; off >>= 1) {
        float ov = __shfl_xor(best, off, 32);
        int   oi = __shfl_xor(bi, off, 32);
        if (ov > best || (ov == best && oi < bi)) { best = ov; bi = oi; }
      }
      if (lane == 0) orow[k] = bi;
      if (lane == (bi & 31)) drow[bi] = (h16)(-65504.f);  // owner invalidates
    }
  }
}

// --------------------------- small support kernels -------------------------
__global__ void prep_kernel(const float* __restrict__ x, h16* __restrict__ xcl,
                            h16* __restrict__ kA, float* __restrict__ xx) {
  int t = blockIdx.x * blockDim.x + threadIdx.x;
  if (t >= BN) return;
  int n = t & (NN - 1);
  int b = t >> 12;
  const float* xb = x + (size_t)b * 8 * NN;
  h16* o = xcl + (size_t)t * 8;
#pragma unroll
  for (int c = 0; c < 8; ++c) o[c] = (h16)xb[(size_t)c * NN + n];
  float v6 = xb[6 * NN + n], v7 = xb[7 * NN + n];
  h16* ka = kA + (size_t)t * 32;
  ka[0] = (h16)v6; ka[1] = (h16)v7;
  for (int c = 2; c < 32; ++c) ka[c] = (h16)0.f;
  xx[t] = v6 * v6 + v7 * v7;
}

__global__ void wcvt_kernel(const float* __restrict__ W, h16* __restrict__ Wh,
                            int Cin, int ldw, long total) {
  long t = (long)blockIdx.x * blockDim.x + threadIdx.x;
  if (t >= total) return;
  int  c = (int)(t % ldw);
  long o = t / ldw;
  Wh[t] = (c < Cin) ? (h16)W[o * Cin + c] : (h16)0.f;
}

__global__ void sumsq_kernel(const h16* __restrict__ F, int ldf, int C,
                             float* __restrict__ xx) {
  int t = blockIdx.x * blockDim.x + threadIdx.x;
  if (t >= BN) return;
  const h16* p = F + (size_t)t * ldf;
  float s = 0.f;
  for (int c = 0; c < C; ++c) { float v = (float)p[c]; s += v * v; }
  xx[t] = s;
}

template <int C, int LDE>
__global__ void gather_kernel(const h16* __restrict__ F, int ldf,
                              const int* __restrict__ idx,
                              h16* __restrict__ E) {
  long t = (long)blockIdx.x * blockDim.x + threadIdx.x;
  if (t >= BNK) return;
  long rn = t >> 5;            // point row (b*N+n); idx layout matches t
  int  b  = (int)(rn >> 12);
  const h16* ctr = F + (size_t)rn * ldf;
  int j = idx[t];
  const h16* nb = F + ((size_t)b * NN + j) * ldf;
  h16* e = E + (size_t)t * LDE;
#pragma unroll
  for (int c = 0; c < C; ++c) {
    e[c]     = nb[c] - ctr[c];
    e[C + c] = ctr[c];
  }
#pragma unroll
  for (int c = 2 * C; c < LDE; ++c) e[c] = (h16)0.f;
}

// max over k: 8 contiguous channels per thread, v8h wide accesses
__global__ void maxk_kernel(const h16* __restrict__ G, h16* __restrict__ out,
                            int ldo) {
  long t = (long)blockIdx.x * blockDim.x + threadIdx.x;
  if (t >= (long)BN * 8) return;
  int  c8 = (int)(t & 7);
  long r  = t >> 3;
  const h16* p = G + (size_t)(r * KK) * 64 + c8 * 8;
  float m[8];
#pragma unroll
  for (int i = 0; i < 8; ++i) m[i] = -3.0e38f;
  for (int k = 0; k < KK; ++k) {
    v8h v = *(const v8h*)(p + (size_t)k * 64);
#pragma unroll
    for (int i = 0; i < 8; ++i) { float f = (float)v[i]; if (f > m[i]) m[i] = f; }
  }
  v8h o;
#pragma unroll
  for (int i = 0; i < 8; ++i) o[i] = (h16)m[i];
  *(v8h*)(out + (size_t)r * ldo + c8 * 8) = o;
}

__global__ void gmax_kernel(const h16* __restrict__ G, float* __restrict__ gm) {
  int t = blockIdx.x * blockDim.x + threadIdx.x;
  if (t >= BB * 1024) return;
  int o = t & 1023, b = t >> 10;
  const h16* p = G + ((size_t)b * NN) * 1024 + o;
  float m = -3.0e38f;
  for (int n = 0; n < NN; ++n) { float v = (float)p[(size_t)n * 1024]; if (v > m) m = v; }
  gm[t] = m;
}

__global__ void hcat_kernel(const float* __restrict__ gm,
                            const h16* __restrict__ cat, h16* __restrict__ H) {
  long t = (long)blockIdx.x * blockDim.x + threadIdx.x;
  if (t >= (long)BN * 1216) return;
  int  c = (int)(t % 1216);
  long r = t / 1216;
  int  b = (int)(r >> 12);
  H[t] = (c < 1024) ? (h16)gm[b * 1024 + c] : cat[(size_t)r * 192 + (c - 1024)];
}

__global__ void head_kernel(const h16* __restrict__ Hh,
                            const float* __restrict__ w9,
                            float* __restrict__ out) {
  int r = blockIdx.x * blockDim.x + threadIdx.x;
  if (r >= BN) return;
  int b = r >> 12, n = r & (NN - 1);
  const h16* p = Hh + (size_t)r * 256;
  float l0 = 0.f, l1 = 0.f;
  for (int c = 0; c < 256; ++c) {
    float v = (float)p[c];
    l0 += v * w9[c];
    l1 += v * w9[256 + c];
  }
  float mx = fmaxf(l0, l1);
  float e0 = __expf(l0 - mx), e1 = __expf(l1 - mx);
  float inv = 1.f / (e0 + e1);
  out[((size_t)b * 2 + 0) * NN + n] = e0 * inv;
  out[((size_t)b * 2 + 1) * NN + n] = e1 * inv;
}

// ---------------------------------------------------------------------------
extern "C" void kernel_launch(void* const* d_in, const int* in_sizes, int n_in,
                              void* d_out, int out_size, void* d_ws,
                              size_t ws_size, hipStream_t stream) {
  (void)in_sizes; (void)n_in; (void)out_size; (void)ws_size;
  const float* x = (const float*)d_in[0];
  const float* w[10];
  for (int i = 1; i <= 9; ++i) w[i] = (const float*)d_in[i];
  const float *s[9], *bi[9];
  for (int i = 1; i <= 8; ++i) {
    s[i]  = (const float*)d_in[10 + 2 * (i - 1)];
    bi[i] = (const float*)d_in[11 + 2 * (i - 1)];
  }
  float* out = (float*)d_out;

  char*  ws  = (char*)d_ws;
  size_t off = 0;
  auto alloc = [&](size_t bytes) -> char* {
    char* p = ws + off;
    off += (bytes + 255) & ~(size_t)255;
    return p;
  };
  h16*   xcl  = (h16*)alloc((size_t)BN * 8 * 2);
  h16*   kA1  = (h16*)alloc((size_t)BN * 32 * 2);
  float* xx   = (float*)alloc((size_t)BN * 4);
  int*   idx  = (int*)alloc((size_t)BN * KK * 4);
  h16* w1h = (h16*)alloc((size_t)64 * 32 * 2);
  h16* w2h = (h16*)alloc((size_t)64 * 64 * 2);
  h16* w3h = (h16*)alloc((size_t)64 * 128 * 2);
  h16* w4h = (h16*)alloc((size_t)64 * 64 * 2);
  h16* w5h = (h16*)alloc((size_t)64 * 128 * 2);
  h16* w6h = (h16*)alloc((size_t)1024 * 192 * 2);
  h16* w7h = (h16*)alloc((size_t)512 * 1216 * 2);
  h16* w8h = (h16*)alloc((size_t)256 * 512 * 2);
  h16*   catb = (h16*)alloc((size_t)BN * 192 * 2);
  float* gm   = (float*)alloc((size_t)BB * 1024 * 4);
  h16* E  = (h16*)alloc((size_t)BNK * 128 * 2);  // 128 MB, reused for g6/hcat
  h16* G1 = (h16*)alloc((size_t)BNK * 64 * 2);   // 64 MB, reused for h7
  h16* G2 = (h16*)alloc((size_t)BNK * 64 * 2);   // 64 MB, reused for h8
  h16* g6   = E;                                 // (BN,1024)
  h16* Hbuf = E + (size_t)BN * 1024;             // (BN,1216)
  h16* h7   = G1;                                // (BN,512)
  h16* h8   = G2;                                // (BN,256)

  const int T = 256;
  auto g1 = [&](long n) { return dim3((unsigned)((n + T - 1) / T)); };
  auto gg = [&](long M, int O) {            // blocks: (M/16/8) * (O/16)
    long mblk = (M / 16) / 8;
    return dim3((unsigned)(mblk * (O / 16)));
  };

  // Weight convert / pad to f16
  wcvt_kernel<<<g1(64 * 32), T, 0, stream>>>(w[1], w1h, 16, 32, 64 * 32);
  wcvt_kernel<<<g1(64 * 64), T, 0, stream>>>(w[2], w2h, 64, 64, 64 * 64);
  wcvt_kernel<<<g1(64 * 128), T, 0, stream>>>(w[3], w3h, 128, 128, 64 * 128);
  wcvt_kernel<<<g1(64 * 64), T, 0, stream>>>(w[4], w4h, 64, 64, 64 * 64);
  wcvt_kernel<<<g1(64 * 128), T, 0, stream>>>(w[5], w5h, 128, 128, 64 * 128);
  wcvt_kernel<<<g1(1024 * 192), T, 0, stream>>>(w[6], w6h, 192, 192, 1024 * 192);
  wcvt_kernel<<<g1((long)512 * 1216), T, 0, stream>>>(w[7], w7h, 1216, 1216, (long)512 * 1216);
  wcvt_kernel<<<g1(256 * 512), T, 0, stream>>>(w[8], w8h, 512, 512, 256 * 512);

  prep_kernel<<<g1(BN), T, 0, stream>>>(x, xcl, kA1, xx);

  // ---- stage 1: knn on (ch6,ch7), edge MLP 16->64->64, max_k -> x1
  knn_kernel<<<dim3(BB * (NN / 16)), 256, 0, stream>>>(kA1, 32, 1, xx, idx);
  gather_kernel<8, 32><<<g1(BNK), T, 0, stream>>>(xcl, 8, idx, E);
  gemm_kernel<<<gg(BNK, 64), 256, 0, stream>>>(E, BNK, 32, 1, w1h, 32, s[1], bi[1], G1, 64, 64, 1);
  gemm_kernel<<<gg(BNK, 64), 256, 0, stream>>>(G1, BNK, 64, 2, w2h, 64, s[2], bi[2], G2, 64, 64, 1);
  maxk_kernel<<<g1((long)BN * 8), T, 0, stream>>>(G2, catb + 0, 192);

  // ---- stage 2: knn on x1 (64d), edge MLP 128->64->64, max_k -> x2
  sumsq_kernel<<<g1(BN), T, 0, stream>>>(catb + 0, 192, 64, xx);
  knn_kernel<<<dim3(BB * (NN / 16)), 256, 0, stream>>>(catb + 0, 192, 2, xx, idx);
  gather_kernel<64, 128><<<g1(BNK), T, 0, stream>>>(catb + 0, 192, idx, E);
  gemm_kernel<<<gg(BNK, 64), 256, 0, stream>>>(E, BNK, 128, 4, w3h, 128, s[3], bi[3], G1, 64, 64, 1);
  gemm_kernel<<<gg(BNK, 64), 256, 0, stream>>>(G1, BNK, 64, 2, w4h, 64, s[4], bi[4], G2, 64, 64, 1);
  maxk_kernel<<<g1((long)BN * 8), T, 0, stream>>>(G2, catb + 64, 192);

  // ---- stage 3: knn on x2 (64d), edge MLP 128->64, max_k -> x3
  sumsq_kernel<<<g1(BN), T, 0, stream>>>(catb + 64, 192, 64, xx);
  knn_kernel<<<dim3(BB * (NN / 16)), 256, 0, stream>>>(catb + 64, 192, 2, xx, idx);
  gather_kernel<64, 128><<<g1(BNK), T, 0, stream>>>(catb + 64, 192, idx, E);
  gemm_kernel<<<gg(BNK, 64), 256, 0, stream>>>(E, BNK, 128, 4, w5h, 128, s[5], bi[5], G1, 64, 64, 1);
  maxk_kernel<<<g1((long)BN * 8), T, 0, stream>>>(G1, catb + 128, 192);

  // ---- global feature + pointwise MLPs + head
  gemm_kernel<<<gg(BN, 1024), 256, 0, stream>>>(catb, BN, 192, 6, w6h, 192, s[6], bi[6], g6, 1024, 1024, 1);
  gmax_kernel<<<g1(BB * 1024), T, 0, stream>>>(g6, gm);
  hcat_kernel<<<g1((long)BN * 1216), T, 0, stream>>>(gm, catb, Hbuf);
  gemm_kernel<<<gg(BN, 512), 256, 0, stream>>>(Hbuf, BN, 1216, 38, w7h, 1216, s[7], bi[7], h7, 512, 512, 1);
  gemm_kernel<<<gg(BN, 256), 256, 0, stream>>>(h7, BN, 512, 16, w8h, 512, s[8], bi[8], h8, 256, 256, 1);
  head_kernel<<<g1(BN), T, 0, stream>>>(h8, w[9], out);
}
